// CausalSelfAttention_68109591380560
// MI455X (gfx1250) — compile-verified, matrix operand order checked by reading, not run on hardware
//
#include <hip/hip_runtime.h>
#include <hip/hip_bf16.h>

typedef __bf16 bf16;
typedef __attribute__((ext_vector_type(16))) __bf16 v16bf;
typedef __attribute__((ext_vector_type(8)))  __bf16 v8bf;
typedef __attribute__((ext_vector_type(8)))  float  v8f;
typedef __attribute__((ext_vector_type(4)))  int    v4i;
typedef __attribute__((ext_vector_type(8)))  int    v8i;
typedef __attribute__((ext_vector_type(4)))  unsigned v4u;

typedef __attribute__((address_space(1))) v4i gv4i;   // global int4
typedef __attribute__((address_space(3))) v4i lv4i;   // LDS int4

#if __has_builtin(__builtin_amdgcn_global_load_async_to_lds_b128)
#define ASYNC_LDS 1
#else
#define ASYNC_LDS 0
#endif

#if __has_builtin(__builtin_amdgcn_tensor_load_to_lds) && __has_builtin(__builtin_amdgcn_s_wait_tensorcnt)
#define HAVE_TDM 1
#else
#define HAVE_TDM 0
#endif

// 16-byte global -> LDS copy. Async (ASYNCcnt-tracked, no VGPR round trip)
// when the gfx1250 builtin is available, else synchronous fallback.
__device__ __forceinline__ void cp16(bf16* l, const bf16* g) {
#if ASYNC_LDS
    __builtin_amdgcn_global_load_async_to_lds_b128(
        (gv4i*)(uintptr_t)g, (lv4i*)(uintptr_t)l, 0, 0);
#else
    *(v8bf*)l = *(const v8bf*)g;
#endif
}

__device__ __forceinline__ void wait_async() {
#if ASYNC_LDS
#if __has_builtin(__builtin_amdgcn_s_wait_asynccnt)
    __builtin_amdgcn_s_wait_asynccnt(0);
#else
    asm volatile("s_wait_asynccnt 0x0" ::: "memory");
#endif
#endif
}

#if HAVE_TDM
// TDM: DMA a 2-D bf16 tile [rows x cols] (global row stride gstride elems)
// into LDS, padding each 128-byte row by 16 bytes (row stride 72 elems).
// cols must be 64 (pad_interval fixed for 128B rows). Issue from one wave.
__device__ __forceinline__ void tdm_load_2d(const bf16* g, unsigned lds_byte,
                                            int rows, int cols, int gstride) {
    unsigned long long ga = (unsigned long long)(uintptr_t)g;
    v4u g0;
    g0[0] = 1u;                                  // count=1 (valid user D#)
    g0[1] = lds_byte;                            // lds_addr (bytes)
    g0[2] = (unsigned)ga;                        // global_addr[31:0]
    g0[3] = (unsigned)(ga >> 32) | (2u << 30);   // global_addr[56:32] | type=2
    v8i g1;
    g1[0] = (1 << 16)        // data_size: 2 bytes
          | (1 << 20)        // pad_enable
          | (4 << 22)        // pad_interval: 32 DWORDs (=128B) between pads
          | (3 << 25);       // pad_amount: 4 DWORDs (=16B)
    g1[1] = (cols & 0xFFFF) << 16;   // tensor_dim0[15:0]   (bits 63:48)
    g1[2] = (rows & 0xFFFF) << 16;   // tensor_dim1[15:0]   (bits 95:80)
    g1[3] = (cols & 0xFFFF) << 16;   // tile_dim0           (bits 127:112)
    g1[4] = (rows & 0xFFFF);         // tile_dim1           (bits 143:128)
    g1[5] = gstride;                 // tensor_dim0_stride[31:0] (elems)
    g1[6] = 0;
    g1[7] = 0;
    v4i z4 = {0, 0, 0, 0};
    v8i z8 = {0, 0, 0, 0, 0, 0, 0, 0};
    __builtin_amdgcn_tensor_load_to_lds(g0, g1, z4, z4, z8, 0);
}
#endif

union FragU { v16bf v; v8bf h8[2]; };

// Load a 16x32 bf16 A/B fragment (wmma_f32_16x16x32_bf16 layout) from a
// row-major tile in LDS. Lane L: row = L&15, half h = L>>4.
// v16bf halves = two 16B chunks at k = kcol+8h and k = kcol+16+8h.
__device__ __forceinline__ v16bf ld_frag(const bf16* base, int ld, int lane, int kcol) {
    int m = lane & 15, h = lane >> 4;
    const bf16* p = base + m * ld + kcol + 8 * h;
    FragU f;
    f.h8[0] = *(const v8bf*)(p);
    f.h8[1] = *(const v8bf*)(p + 16);
    return f.v;
}

__device__ __forceinline__ v8f wmma_bf16(v16bf a, v16bf b, v8f c) {
    return __builtin_amdgcn_wmma_f32_16x16x32_bf16(false, a, false, b, (short)0, c, false, false);
}

__global__ void f32_to_bf16_kernel(const float* __restrict__ in, bf16* __restrict__ out, int n) {
    int i = blockIdx.x * blockDim.x + threadIdx.x;
    int stride = gridDim.x * blockDim.x;
    for (; i < n; i += stride) out[i] = (bf16)in[i];
}

// C[m][n] = sum_k A[m][k] * W[n][k] + bias[n]   (torch Linear: x @ W^T + b)
// A: [M,K] bf16 row-major, W: [N,K] bf16 row-major.
// BM=128 BN=128 BK=32, 256 threads = 8 waves (4x2), wave block 32x64 (2x4 WMMA
// tiles). Double-buffered async global->LDS staging.
template<bool OUT_BF16>
__global__ __launch_bounds__(256) void gemm_bias_kernel(
    const bf16* __restrict__ A, const bf16* __restrict__ W,
    const float* __restrict__ bias, void* __restrict__ Out,
    int M, int N, int K) {
    constexpr int BM = 128, BN = 128, BK = 32, LDT = BK + 8;
    __shared__ bf16 As[2][BM * LDT];
    __shared__ bf16 Bs[2][BN * LDT];
    const int tid = threadIdx.x;
    const int lane = tid & 31, wave = tid >> 5;
    const int wr = wave & 3, wc = wave >> 2;       // 4x2 waves: 32 rows x 64 cols
    const int bm = blockIdx.x * BM;
    const int bn = blockIdx.y * BN;

    // Staging: 2 threads per row, 16 elems (2x16B) each, for both A and B.
    const int row = tid >> 1, ch = (tid & 1) * 16;
    const bf16* ga = A + (size_t)(bm + row) * K + ch;
    const bf16* gb = W + (size_t)(bn + row) * K + ch;

    v8f acc[2][4] = {};

    cp16(&As[0][row * LDT + ch],     ga);
    cp16(&As[0][row * LDT + ch + 8], ga + 8);
    cp16(&Bs[0][row * LDT + ch],     gb);
    cp16(&Bs[0][row * LDT + ch + 8], gb + 8);
    wait_async();
    __syncthreads();

    int buf = 0;
    for (int k0 = 0; k0 < K; k0 += BK) {
        const int nxt = k0 + BK;
        if (nxt < K) {   // issue next tile's async copies into the other buffer
            cp16(&As[buf ^ 1][row * LDT + ch],     ga + nxt);
            cp16(&As[buf ^ 1][row * LDT + ch + 8], ga + nxt + 8);
            cp16(&Bs[buf ^ 1][row * LDT + ch],     gb + nxt);
            cp16(&Bs[buf ^ 1][row * LDT + ch + 8], gb + nxt + 8);
        }
        v16bf a0 = ld_frag(&As[buf][(wr * 32)      * LDT], LDT, lane, 0);
        v16bf a1 = ld_frag(&As[buf][(wr * 32 + 16) * LDT], LDT, lane, 0);
        #pragma unroll
        for (int nt = 0; nt < 4; ++nt) {
            v16bf bfr = ld_frag(&Bs[buf][(wc * 64 + nt * 16) * LDT], LDT, lane, 0);
            acc[0][nt] = wmma_bf16(a0, bfr, acc[0][nt]);
            acc[1][nt] = wmma_bf16(a1, bfr, acc[1][nt]);
        }
        wait_async();
        __syncthreads();
        buf ^= 1;
    }

    const int nn = lane & 15, hh = lane >> 4;
    const int m0 = bm + wr * 32, n0 = bn + wc * 64;
    #pragma unroll
    for (int mt = 0; mt < 2; ++mt) {
        #pragma unroll
        for (int nt = 0; nt < 4; ++nt) {
            int col = n0 + nt * 16 + nn;
            float bv = bias[col];
            #pragma unroll
            for (int r = 0; r < 8; ++r) {
                int rw = m0 + mt * 16 + r + 8 * hh;
                float v = acc[mt][nt][r] + bv;
                if (OUT_BF16) ((bf16*)Out)[(size_t)rw * N + col] = (bf16)v;
                else          ((float*)Out)[(size_t)rw * N + col] = v;
            }
        }
    }
}

// Flash attention, causal. qkv: [B*T, 3C] bf16 (q|k|v each C). y: [B*T, C] bf16.
// Block = (query tile of 64 rows) x (b,h). 128 threads = 4 waves; wave w owns
// query rows [16w, 16w+16) of the tile. O accum 16x64 f32 per wave (4 v8f).
// Q/K tiles are DMA'd into LDS by the Tensor Data Mover (padded rows),
// V is transposed into LDS manually (TDM has no transpose).
__global__ __launch_bounds__(128) void flash_attn_kernel(
    const bf16* __restrict__ qkv, bf16* __restrict__ y) {
    constexpr int T = 2048, C = 1024, H = 16, Dh = 64, BQ = 64, BKV = 64;
    constexpr int LD = Dh + 8;      // padded LDS row (bf16 elems), 16B-aligned rows
    __shared__ bf16 Qs[BQ * LD];
    __shared__ bf16 Ks[BKV * LD];
    __shared__ bf16 Vts[Dh * LD];          // V transposed: Vts[d][t]
    __shared__ bf16 Ps[4 * 16 * LD];       // per-wave P staging (C-layout -> A-layout)

    const int tid = threadIdx.x, lane = tid & 31, wave = tid >> 5;
    const int nn = lane & 15, hh = lane >> 4;
    const int qblk = blockIdx.x;                 // 0..T/BQ-1
    const int b = blockIdx.y / H, h = blockIdx.y % H;
    const int qbase = qblk * BQ;
    const size_t rs = 3 * C;                     // qkv row stride (elems)
    const bf16* qp = qkv + (size_t)b * T * rs + h * Dh;
    const bf16* kp = qp + C;
    const bf16* vp = qp + 2 * C;

    const int lrow = tid >> 1, lch = (tid & 1) * 32;  // 2 thr/row, 32 elems each

    // Load Q tile (row-major, padded) — TDM if available.
#if HAVE_TDM
    if (wave == 0)
        tdm_load_2d(qp + (size_t)qbase * rs, (unsigned)(uintptr_t)Qs, BQ, Dh, (int)rs);
#else
    {
        const bf16* g = qp + (size_t)(qbase + lrow) * rs + lch;
        #pragma unroll
        for (int j = 0; j < 4; ++j)
            cp16(&Qs[lrow * LD + lch + j * 8], g + j * 8);
    }
#endif

    v8f accO[4] = {};
    float m_i[8], l_i[8];
    #pragma unroll
    for (int r = 0; r < 8; ++r) { m_i[r] = -3.0e38f; l_i[r] = 0.0f; }
    const float scale = 0.125f;                      // 1/sqrt(Dh)
    const float LOG2E = 1.4426950408889634f;

    for (int kt = 0; kt <= qblk; ++kt) {
        const int kbase = kt * BKV;
        __syncthreads();   // previous tile's readers done before overwrite
#if HAVE_TDM
        if (wave == 0)
            tdm_load_2d(kp + (size_t)kbase * rs, (unsigned)(uintptr_t)Ks, BKV, Dh, (int)rs);
#else
        {
            const bf16* gk = kp + (size_t)(kbase + lrow) * rs + lch;
            #pragma unroll
            for (int j = 0; j < 4; ++j)
                cp16(&Ks[lrow * LD + lch + j * 8], gk + j * 8);
        }
#endif
        {   // V, transposed into LDS: Vts[d][t] = V[t][d]
            const bf16* gv = vp + (size_t)(kbase + lrow) * rs + lch;
            #pragma unroll
            for (int j = 0; j < 32; ++j)
                Vts[(lch + j) * LD + lrow] = gv[j];
        }
#if HAVE_TDM
        if (wave == 0) __builtin_amdgcn_s_wait_tensorcnt(0);
#else
        wait_async();
#endif
        __syncthreads();

        // S = (Q K^T) * scale   — 16x64 per wave = 4 n-tiles x 2 k-steps
        v8f accS[4] = {};
        #pragma unroll
        for (int ks = 0; ks < Dh; ks += 32) {
            v16bf aq = ld_frag(&Qs[(wave * 16) * LD], LD, lane, ks);
            #pragma unroll
            for (int nt = 0; nt < 4; ++nt) {
                v16bf bk = ld_frag(&Ks[(nt * 16) * LD], LD, lane, ks);
                accS[nt] = wmma_bf16(aq, bk, accS[nt]);
            }
        }

        // Scale, causal mask (diagonal tile only), row max.
        float rowmax[8];
        #pragma unroll
        for (int r = 0; r < 8; ++r) rowmax[r] = -3.0e38f;
        #pragma unroll
        for (int nt = 0; nt < 4; ++nt) {
            #pragma unroll
            for (int r = 0; r < 8; ++r) {
                float s = accS[nt][r] * scale;
                if (kt == qblk) {
                    int q  = qbase + wave * 16 + r + 8 * hh;
                    int kk = kbase + nt * 16 + nn;
                    if (kk > q) s = -3.0e38f;
                }
                accS[nt][r] = s;
                rowmax[r] = fmaxf(rowmax[r], s);
            }
        }
        // Row m = r + 8*hh lives in one 16-lane half -> xor masks 1,2,4,8.
        #pragma unroll
        for (int r = 0; r < 8; ++r) {
            float v = rowmax[r];
            v = fmaxf(v, __shfl_xor(v, 1, 32));
            v = fmaxf(v, __shfl_xor(v, 2, 32));
            v = fmaxf(v, __shfl_xor(v, 4, 32));
            v = fmaxf(v, __shfl_xor(v, 8, 32));
            rowmax[r] = v;
        }
        float mnew[8], alpha[8], rowsum[8];
        #pragma unroll
        for (int r = 0; r < 8; ++r) {
            mnew[r]  = fmaxf(m_i[r], rowmax[r]);
            alpha[r] = exp2f((m_i[r] - mnew[r]) * LOG2E);
            rowsum[r] = 0.0f;
        }

        // P = exp(S - mnew): stage bf16 into per-wave LDS (A-layout source).
        bf16* pw = &Ps[wave * 16 * LD];
        #pragma unroll
        for (int nt = 0; nt < 4; ++nt) {
            #pragma unroll
            for (int r = 0; r < 8; ++r) {
                float p = exp2f((accS[nt][r] - mnew[r]) * LOG2E);
                rowsum[r] += p;
                pw[(r + 8 * hh) * LD + nt * 16 + nn] = (bf16)p;
            }
        }
        #pragma unroll
        for (int r = 0; r < 8; ++r) {
            float v = rowsum[r];
            v += __shfl_xor(v, 1, 32);
            v += __shfl_xor(v, 2, 32);
            v += __shfl_xor(v, 4, 32);
            v += __shfl_xor(v, 8, 32);
            l_i[r] = l_i[r] * alpha[r] + v;
            m_i[r] = mnew[r];
        }
        #pragma unroll
        for (int nt = 0; nt < 4; ++nt)
            #pragma unroll
            for (int r = 0; r < 8; ++r) accO[nt][r] *= alpha[r];

        // O += P @ V   (B-frags read rows of Vts = columns of V)
        #pragma unroll
        for (int ks = 0; ks < BKV; ks += 32) {
            v16bf ap = ld_frag(pw, LD, lane, ks);
            #pragma unroll
            for (int nt = 0; nt < 4; ++nt) {
                v16bf bv = ld_frag(&Vts[(nt * 16) * LD], LD, lane, ks);
                accO[nt] = wmma_bf16(ap, bv, accO[nt]);
            }
        }
    }

    // y = O / l
    #pragma unroll
    for (int nt = 0; nt < 4; ++nt) {
        #pragma unroll
        for (int r = 0; r < 8; ++r) {
            int q = qbase + wave * 16 + r + 8 * hh;
            int d = nt * 16 + nn;
            float ov = accO[nt][r] / l_i[r];
            y[((size_t)b * T + q) * C + h * Dh + d] = (bf16)ov;
        }
    }
}

extern "C" void kernel_launch(void* const* d_in, const int* in_sizes, int n_in,
                              void* d_out, int out_size, void* d_ws, size_t ws_size,
                              hipStream_t stream) {
    const float* x     = (const float*)d_in[0];
    const float* qkv_w = (const float*)d_in[1];
    const float* qkv_b = (const float*)d_in[2];
    const float* out_w = (const float*)d_in[3];
    const float* out_b = (const float*)d_in[4];
    float* out = (float*)d_out;

    const int B = 4, T = 2048, C = 1024;
    const int M = B * T;          // 8192
    const int N3 = 3 * C;         // 3072

    // Workspace layout (bf16), ~92 MB total.
    bf16* xb  = (bf16*)d_ws;                   // [M, C]
    bf16* wqb = xb  + (size_t)M * C;           // [3C, C]
    bf16* wob = wqb + (size_t)N3 * C;          // [C, C]
    bf16* qkv = wob + (size_t)C * C;           // [M, 3C]
    bf16* yb  = qkv + (size_t)M * N3;          // [M, C]

    f32_to_bf16_kernel<<<2048, 256, 0, stream>>>(x,     xb,  M * C);
    f32_to_bf16_kernel<<<2048, 256, 0, stream>>>(qkv_w, wqb, N3 * C);
    f32_to_bf16_kernel<<<1024, 256, 0, stream>>>(out_w, wob, C * C);

    dim3 g1(M / 128, N3 / 128);
    gemm_bias_kernel<true><<<g1, 256, 0, stream>>>(xb, wqb, qkv_b, qkv, M, N3, C);

    dim3 g2(T / 64, B * 16);
    flash_attn_kernel<<<g2, 128, 0, stream>>>(qkv, yb);

    dim3 g3(M / 128, C / 128);
    gemm_bias_kernel<false><<<g3, 256, 0, stream>>>(yb, wob, out_b, out, M, C, C);
}
